// FP8LinearDeepGEMM_33243046871361
// MI455X (gfx1250) — 8-bit matrix operand order repaired and checked by reading, compiled, not run on hardware
//
#include <hip/hip_runtime.h>
#include <stdint.h>

typedef __attribute__((ext_vector_type(16))) int   v16i;
typedef __attribute__((ext_vector_type(8)))  float v8f;
typedef __attribute__((ext_vector_type(4)))  float v4f;
typedef __attribute__((ext_vector_type(4)))  int   v4i;

typedef v4i __attribute__((address_space(1)))* v4i_gptr;
typedef v4i __attribute__((address_space(3)))* v4i_lptr;

#define GM 8192
#define GK 4096
#define GN 4096
#define KT 32
#define FP8_MAX 448.0f
#define BSTRIDE 144

#if defined(__HIP_DEVICE_COMPILE__) && \
    __has_builtin(__builtin_amdgcn_global_load_async_to_lds_b128) && \
    __has_builtin(__builtin_amdgcn_s_wait_asynccnt)
#define USE_ASYNC_LDS 1
#else
#define USE_ASYNC_LDS 0
#endif

__device__ __forceinline__ void stage_b128(const uint8_t* __restrict__ g, uint8_t* l) {
#if USE_ASYNC_LDS
  __builtin_amdgcn_global_load_async_to_lds_b128((v4i_gptr)g, (v4i_lptr)l, 0, 0);
#else
  *(v4i*)l = *(const v4i*)g;
#endif
}

__device__ __forceinline__ unsigned f32_to_e4m3(float f) {
  unsigned u    = __float_as_uint(f);
  unsigned sign = (u >> 24) & 0x80u;
  unsigned a    = u & 0x7fffffffu;
  if (a >= 0x43e00000u) {
    return sign | ((a > 0x7f800000u) ? 0x7fu : 0x7eu);
  }
  if (a < 0x3c800000u) {
    float av = __uint_as_float(a);
    unsigned q = (unsigned)(int)rintf(av * 512.0f);
    return sign | q;
  }
  unsigned r = a + 0x7ffffu + ((a >> 20) & 1u);
  return sign | ((r >> 20) - 960u);
}

__device__ __forceinline__ unsigned pack_e4m3_4(float a, float b, float c, float d) {
  return f32_to_e4m3(a) | (f32_to_e4m3(b) << 8) |
         (f32_to_e4m3(c) << 16) | (f32_to_e4m3(d) << 24);
}

__global__ __launch_bounds__(256) void quant_act_1x128(
    const float* __restrict__ x, uint8_t* __restrict__ xq, float* __restrict__ xs) {
  const int wave = threadIdx.x >> 5;
  const int lane = threadIdx.x & 31;
  const size_t tile = (size_t)blockIdx.x * 8 + wave;
  const size_t base = tile * 128;

  v4f v = *(const v4f*)(x + base + (size_t)lane * 4);
  float am = fmaxf(fmaxf(fabsf(v[0]), fabsf(v[1])), fmaxf(fabsf(v[2]), fabsf(v[3])));
#pragma unroll
  for (int off = 16; off > 0; off >>= 1)
    am = fmaxf(am, __shfl_xor(am, off, 32));

  float safe = fmaxf(am, 1e-4f);
  float mult = FP8_MAX / safe;
  unsigned packed = pack_e4m3_4(v[0] * mult, v[1] * mult, v[2] * mult, v[3] * mult);
  unsigned dst = ((lane >> 2) & 1) * 64 + (lane >> 3) * 16 + (lane & 3) * 4;
  *(unsigned*)(xq + base + dst) = packed;
  if (lane == 0) xs[tile] = safe / FP8_MAX;
}

__global__ __launch_bounds__(256) void quant_weight_128x128(
    const float* __restrict__ w, uint8_t* __restrict__ wq, float* __restrict__ wsc) {
  __shared__ float red[256];
  const int kb = blockIdx.x;
  const int nb = blockIdx.y;
  const int t  = threadIdx.x;

  float am = 0.0f;
#pragma unroll 4
  for (int it = 0; it < 16; ++it) {
    int f   = t + it * 256;
    int row = f >> 5;
    int c4  = f & 31;
    size_t off = (size_t)(nb * 128 + row) * GK + (size_t)kb * 128 + (size_t)c4 * 4;
    v4f v = *(const v4f*)(w + off);
    am = fmaxf(am, fmaxf(fmaxf(fabsf(v[0]), fabsf(v[1])),
                         fmaxf(fabsf(v[2]), fabsf(v[3]))));
  }
  red[t] = am;
  __syncthreads();
  for (int s = 128; s > 0; s >>= 1) {
    if (t < s) red[t] = fmaxf(red[t], red[t + s]);
    __syncthreads();
  }
  float safe = fmaxf(red[0], 1e-6f);
  float mult = FP8_MAX / safe;
  if (t == 0) wsc[nb * KT + kb] = safe / FP8_MAX;

#pragma unroll 4
  for (int it = 0; it < 16; ++it) {
    int f   = t + it * 256;
    int row = f >> 5;
    int c4  = f & 31;
    size_t off = (size_t)(nb * 128 + row) * GK + (size_t)kb * 128 + (size_t)c4 * 4;
    v4f v = *(const v4f*)(w + off);
    *(unsigned*)(wq + off) =
        pack_e4m3_4(v[0] * mult, v[1] * mult, v[2] * mult, v[3] * mult);
  }
}

__global__ __launch_bounds__(256) void fp8_gemm_scaled(
    const uint8_t* __restrict__ xq, const float* __restrict__ xs,
    const uint8_t* __restrict__ wq, const float* __restrict__ wsc,
    float* __restrict__ out) {
  __shared__ float   s_as[128 * KT];
  __shared__ uint8_t s_b[2][128 * BSTRIDE];

  const int t    = threadIdx.x;
  const int lane = t & 31;
  const int wave = t >> 5;
  const int hi   = lane >> 4;
  const int lp   = lane & 15;

  const int m_wg   = blockIdx.y * 128;
  const int n_wg   = blockIdx.x * 128;
  const int nb     = blockIdx.x;
  const int m_wave = m_wg + (wave >> 1) * 32;
  const int n_loc  = (wave & 1) * 64;

  {
    const int row = t >> 1, half = (t & 1) * 16;
    const float* g = xs + (size_t)(m_wg + row) * KT + half;
    float* l = s_as + row * KT + half;
#pragma unroll
    for (int j = 0; j < 16; j += 4)
      *(v4f*)(l + j) = *(const v4f*)(g + j);
  }

  auto stageB = [&](int kt, int buf) {
#pragma unroll
    for (int i = 0; i < 4; ++i) {
      int ch  = t + 256 * i;
      int row = ch >> 3;
      int col = (ch & 7) * 16;
      const uint8_t* g = wq + (size_t)(n_wg + row) * GK + (size_t)kt * 128 + col;
      stage_b128(g, &s_b[buf][row * BSTRIDE + col]);
    }
  };

  stageB(0, 0);

  v8f acc[2][4];
#pragma unroll
  for (int mt = 0; mt < 2; ++mt)
#pragma unroll
    for (int nt = 0; nt < 4; ++nt)
      acc[mt][nt] = (v8f){0.f, 0.f, 0.f, 0.f, 0.f, 0.f, 0.f, 0.f};

  for (int kt = 0; kt < KT; ++kt) {
    const int buf = kt & 1;
    if (kt + 1 < KT) {
      stageB(kt + 1, buf ^ 1);
#if USE_ASYNC_LDS
      __builtin_amdgcn_s_wait_asynccnt(4);
#endif
    } else {
#if USE_ASYNC_LDS
      __builtin_amdgcn_s_wait_asynccnt(0);
#endif
    }
    __syncthreads();

    const float sw = wsc[nb * KT + kt];

    v16i a[2];
#pragma unroll
    for (int mt = 0; mt < 2; ++mt) {
      const size_t row = (size_t)(m_wave + mt * 16 + lp);
      const v4i* p = (const v4i*)(xq + row * GK + (size_t)kt * 128 + hi * 64);
#pragma unroll
      for (int c = 0; c < 4; ++c) {
        v4i q = p[c];
        a[mt][4 * c + 0] = q[0];
        a[mt][4 * c + 1] = q[1];
        a[mt][4 * c + 2] = q[2];
        a[mt][4 * c + 3] = q[3];
      }
    }

    float sa[2][8];
#pragma unroll
    for (int mt = 0; mt < 2; ++mt)
#pragma unroll
      for (int r = 0; r < 8; ++r)
        sa[mt][r] = s_as[((m_wave - m_wg) + mt * 16 + 8 * hi + r) * KT + kt] * sw;

    const v8f zero = (v8f){0.f, 0.f, 0.f, 0.f, 0.f, 0.f, 0.f, 0.f};
#pragma unroll
    for (int nt = 0; nt < 4; ++nt) {
      v16i b;
      const uint8_t* lb = &s_b[buf][(n_loc + nt * 16 + lp) * BSTRIDE + 16 * hi];
#pragma unroll
      for (int g = 0; g < 4; ++g) {
        v4i q = *(const v4i*)(lb + 32 * g);
        b[4 * g + 0] = q[0];
        b[4 * g + 1] = q[1];
        b[4 * g + 2] = q[2];
        b[4 * g + 3] = q[3];
      }
#pragma unroll
      for (int mt = 0; mt < 2; ++mt) {
        v8f p = __builtin_amdgcn_wmma_f32_16x16x128_fp8_fp8(
            a[mt], b, (short)0, zero, false, false);
#pragma unroll
        for (int r = 0; r < 8; ++r)
          acc[mt][nt][r] += p[r] * sa[mt][r];
      }
    }
    __syncthreads();
  }

#pragma unroll
  for (int mt = 0; mt < 2; ++mt)
#pragma unroll
    for (int nt = 0; nt < 4; ++nt)
#pragma unroll
      for (int r = 0; r < 8; ++r) {
        size_t m = (size_t)(m_wave + mt * 16 + 8 * hi + r);
        size_t n = (size_t)(n_wg + n_loc + nt * 16 + lp);
        out[m * GN + n] = acc[mt][nt][r];
      }
}

extern "C" void kernel_launch(void* const* d_in, const int* in_sizes, int n_in,
                              void* d_out, int out_size, void* d_ws, size_t ws_size,
                              hipStream_t stream) {
  const float* x = (const float*)d_in[0];
  const float* w = (const float*)d_in[1];
  float* out = (float*)d_out;

  uint8_t* ws8 = (uint8_t*)d_ws;
  uint8_t* xq  = ws8;
  uint8_t* wq  = ws8 + (size_t)GM * GK;
  float*   xs  = (float*)(ws8 + (size_t)GM * GK + (size_t)GN * GK);
  float*   wsc = xs + (size_t)GM * KT;

  quant_act_1x128<<<(GM * KT) / 8, 256, 0, stream>>>(x, xq, xs);

  dim3 gw(GK / 128, GN / 128);
  quant_weight_128x128<<<gw, 256, 0, stream>>>(w, wq, wsc);

  dim3 gg(GN / 128, GM / 128);
  fp8_gemm_scaled<<<gg, 256, 0, stream>>>(xq, xs, wq, wsc, out);

  (void)in_sizes; (void)n_in; (void)out_size; (void)ws_size;
}
